// NAMLSSV2Rebuilt_29171417874737
// MI455X (gfx1250) — matrix operand order repaired and checked by reading, not compiled, not run on hardware
//
#include <hip/hip_runtime.h>
#include <hip/hip_bf16.h>
#include <math.h>

// Problem dims (fixed by the reference)
#define BB   16384
#define FF   64
#define NB   64
#define H1   128
#define H2   128
#define KK   3

#define BM   64          // batch rows per workgroup
#define SA_STRIDE 136    // bf16 elements per A-stage row (64|128 + pad, keeps 16B align)
#define SH_STRIDE 132    // fp32 elements per staging row (conflict-free with 4-lane LN)

typedef float  v8f  __attribute__((ext_vector_type(8)));
typedef __bf16 v16bf __attribute__((ext_vector_type(16)));
typedef unsigned short v8us  __attribute__((ext_vector_type(8)));
typedef unsigned short v16us __attribute__((ext_vector_type(16)));

__device__ __forceinline__ unsigned short f2bf(float x) {
    union { float f; unsigned u; } v; v.f = x;
    unsigned r = v.u + 0x7FFFu + ((v.u >> 16) & 1u);   // round-to-nearest-even
    return (unsigned short)(r >> 16);
}

__device__ __forceinline__ v8f wmma_bf16(v16bf a, v16bf b, v8f c) {
    return __builtin_amdgcn_wmma_f32_16x16x32_bf16(false, a, false, b, (short)0, c, false, false);
}

// B fragment: 32x16 bf16, lane L -> N=n0+(L&15), K runs {k0+8h..+7, k0+16+8h..+7}. ld = 128.
__device__ __forceinline__ v16bf loadB(const unsigned short* __restrict__ w,
                                       int k0, int n0, int lane) {
    const unsigned short* p = w + (size_t)(k0 + 8 * (lane >> 4)) * 128 + n0 + (lane & 15);
    v16us t;
#pragma unroll
    for (int i = 0; i < 8; ++i) { t[i] = p[i * 128]; t[i + 8] = p[(i + 16) * 128]; }
    return __builtin_bit_cast(v16bf, t);
}

// A fragment: 16x32 bf16 from LDS, lane L -> M=m0+(L&15), K runs {k0+8h..+7, k0+16+8h..+7}
__device__ __forceinline__ v16bf loadA(const unsigned short* __restrict__ sA,
                                       int m0, int k0, int lane) {
    int m = m0 + (lane & 15);
    int h = lane >> 4;
    const unsigned short* p = sA + m * SA_STRIDE + k0 + 8 * h;
    v8us lo = *(const v8us*)p;
    v8us hi = *(const v8us*)(p + 16);
    v16us t;
#pragma unroll
    for (int i = 0; i < 8; ++i) { t[i] = lo[i]; t[i + 8] = hi[i]; }
    return __builtin_bit_cast(v16bf, t);
}

__device__ __forceinline__ float softplusf(float x) {
    return (x > 20.f) ? x : log1pf(__expf(x));
}

// ------- pre-pass: weights fp32 -> bf16, plus reciprocal RBF widths -------
__global__ __launch_bounds__(256) void cvt_weights(
    const float* __restrict__ W1, const float* __restrict__ W2,
    const float* __restrict__ Wr, const float* __restrict__ logw,
    unsigned short* __restrict__ ws, float* __restrict__ rw) {
    const int N1 = FF * NB * H1;          // 524288
    const int N2 = FF * H1 * H2;          // 1048576
    const int NR = FF * NB * H2;          // 524288
    const int NT = N1 + N2 + NR;          // 2097152
    int idx = blockIdx.x * 256 + threadIdx.x;
    if (idx < NT) {
        float v;
        if (idx < N1)            v = W1[idx];
        else if (idx < N1 + N2)  v = W2[idx - N1];
        else                     v = Wr[idx - N1 - N2];
        ws[idx] = f2bf(v);
    } else {
        int i = idx - NT;                 // 0 .. FF*NB-1
        if (i < FF * NB) {
            float lw = logw[i];
            float w  = expf(fminf(fmaxf(lw, -5.f), 5.f)) + 0.1f;  // full-precision once
            rw[i] = 1.f / w;
        }
    }
}

// ---------------- fused NAM kernel ----------------
__global__ __launch_bounds__(256) void nam_main(
    const float* __restrict__ x, const float* __restrict__ centers,
    const float* __restrict__ rw,
    const float* __restrict__ b1, const float* __restrict__ g1, const float* __restrict__ be1,
    const float* __restrict__ b2, const float* __restrict__ g2, const float* __restrict__ be2,
    const float* __restrict__ br, const float* __restrict__ att,
    const float* __restrict__ bias,
    const float* __restrict__ Wpi, const float* __restrict__ bpi,
    const float* __restrict__ Wa,  const float* __restrict__ ba,
    const float* __restrict__ Wb,  const float* __restrict__ bb,
    const unsigned short* __restrict__ w1b, const unsigned short* __restrict__ w2b,
    const unsigned short* __restrict__ wrb, float* __restrict__ out) {

    __shared__ __align__(16) unsigned short sA[BM * SA_STRIDE]; // bf16 A stage (rbf / gelu1)
    __shared__ __align__(16) float sH[BM * SH_STRIDE];          // fp32 staging
    __shared__ float sV[7 * 128];                               // b1,g1,be1,b2,g2,be2,br

    const int tid  = threadIdx.x;
    const int lane = tid & 31;
    const int wv   = tid >> 5;      // wave id == N-block
    const int n0   = wv * 16;
    const int hf   = lane >> 4;     // fragment half
    const int nl   = lane & 15;
    const int n    = n0 + nl;       // this lane's output column
    const int b0   = blockIdx.x * BM;

    const int rowLN = tid >> 2;     // 4 lanes per row for LN / rbf / head
    const int tLN   = tid & 3;

    // softmax(att) normalizers (redundant per thread; 64 L2-cached scalars)
    float amax = -1e30f;
    for (int i = 0; i < FF; ++i) amax = fmaxf(amax, att[i]);
    float asum = 0.f;
    for (int i = 0; i < FF; ++i) asum += __expf(att[i] - amax);

    v8f agg[4] = {{}, {}, {}, {}};

    for (int f = 0; f < FF; ++f) {
        // prefetch next feature's weight slices (64KB = 256 x 256B lines, 1/thread)
        if (f + 1 < FF) {
            const char* p;
            if (tid < 64)        p = (const char*)(w1b + (size_t)(f + 1) * NB * H1) + tid * 256;
            else if (tid < 192)  p = (const char*)(w2b + (size_t)(f + 1) * H1 * H2) + (tid - 64) * 256;
            else                 p = (const char*)(wrb + (size_t)(f + 1) * NB * H2) + (tid - 192) * 256;
            __builtin_prefetch(p, 0, 1);
        }

        __syncthreads();  // prior iteration done with sA/sH/sV

        // stage per-feature bias/gain vectors
        for (int i = tid; i < 7 * 128; i += 256) {
            int a = i >> 7, c = i & 127;
            float v;
            if      (a == 0) v = b1 [f * 128 + c];
            else if (a == 1) v = g1 [f * 128 + c];
            else if (a == 2) v = be1[f * 128 + c];
            else if (a == 3) v = b2 [f * 128 + c];
            else if (a == 4) v = g2 [f * 128 + c];
            else if (a == 5) v = be2[f * 128 + c];
            else             v = br [f * 128 + c];
            sV[i] = v;
        }

        // RBF expansion -> sA (bf16), 16 basis values per thread
        {
            float xv = x[(size_t)(b0 + rowLN) * FF + f];
            xv = fminf(fmaxf(xv, -10.f), 10.f);
            unsigned short tmp[16];
#pragma unroll
            for (int j = 0; j < 16; ++j) {
                int nb = tLN * 16 + j;
                float c   = centers[f * NB + nb];
                float rwv = rw[f * NB + nb];
                float d   = (xv - c) * rwv;
                d = fminf(fmaxf(d, -10.f), 10.f);
                tmp[j] = f2bf(__expf(-0.5f * d * d));
            }
            v8us lo, hi;
#pragma unroll
            for (int j = 0; j < 8; ++j) { lo[j] = tmp[j]; hi[j] = tmp[j + 8]; }
            *(v8us*)&sA[rowLN * SA_STRIDE + tLN * 16]     = lo;
            *(v8us*)&sA[rowLN * SA_STRIDE + tLN * 16 + 8] = hi;
        }
        __syncthreads();

        // GEMM1 (rbf x W1) and GEMMr (rbf x Wr), K = 64
        v8f c1[4] = {{}, {}, {}, {}};
        v8f cr[4] = {{}, {}, {}, {}};
        const unsigned short* w1p = w1b + (size_t)f * NB * H1;
        const unsigned short* wrp = wrb + (size_t)f * NB * H2;
#pragma unroll
        for (int ks = 0; ks < 2; ++ks) {
            int k0 = ks * 32;
            v16bf bW1 = loadB(w1p, k0, n0, lane);
            v16bf bWr = loadB(wrp, k0, n0, lane);
#pragma unroll
            for (int mt = 0; mt < 4; ++mt) {
                v16bf a = loadA(sA, mt * 16, k0, lane);
                c1[mt] = wmma_bf16(a, bW1, c1[mt]);
                cr[mt] = wmma_bf16(a, bWr, cr[mt]);
            }
        }

        // stage h1 = c1 + b1 into sH (fragment mapping)
        {
            float b1v = sV[0 * 128 + n];
#pragma unroll
            for (int mt = 0; mt < 4; ++mt)
#pragma unroll
                for (int r = 0; r < 8; ++r)
                    sH[(mt * 16 + hf * 8 + r) * SH_STRIDE + n] = c1[mt][r] + b1v;
        }
        __syncthreads();

        // LN1 + GELU -> sA (bf16)
        {
            float s = 0.f, ss = 0.f;
#pragma unroll
            for (int i = 0; i < 32; ++i) {
                float v = sH[rowLN * SH_STRIDE + 4 * i + tLN];
                s += v; ss += v * v;
            }
            s  += __shfl_xor(s, 1);  s  += __shfl_xor(s, 2);
            ss += __shfl_xor(ss, 1); ss += __shfl_xor(ss, 2);
            float mean = s * (1.f / 128.f);
            float var  = ss * (1.f / 128.f) - mean * mean;
            float rs   = rsqrtf(var + 1e-5f);
#pragma unroll
            for (int i = 0; i < 32; ++i) {
                int c = 4 * i + tLN;
                float v  = sH[rowLN * SH_STRIDE + c];
                float xn = (v - mean) * rs * sV[1 * 128 + c] + sV[2 * 128 + c];
                float ge = 0.5f * xn * (1.f + erff(xn * 0.70710678118f));
                sA[rowLN * SA_STRIDE + c] = f2bf(ge);
            }
        }
        __syncthreads();

        // GEMM2 (gelu1 x W2), K = 128
#pragma unroll
        for (int mt = 0; mt < 4; ++mt) c1[mt] = (v8f){};
        const unsigned short* w2p = w2b + (size_t)f * H1 * H2;
#pragma unroll
        for (int ks = 0; ks < 4; ++ks) {
            int k0 = ks * 32;
            v16bf bW2 = loadB(w2p, k0, n0, lane);
#pragma unroll
            for (int mt = 0; mt < 4; ++mt) {
                v16bf a = loadA(sA, mt * 16, k0, lane);
                c1[mt] = wmma_bf16(a, bW2, c1[mt]);
            }
        }
        {
            float b2v = sV[3 * 128 + n];
#pragma unroll
            for (int mt = 0; mt < 4; ++mt)
#pragma unroll
                for (int r = 0; r < 8; ++r)
                    sH[(mt * 16 + hf * 8 + r) * SH_STRIDE + n] = c1[mt][r] + b2v;
        }
        __syncthreads();

        // LN2 + GELU, in-place in sH (fp32)
        {
            float s = 0.f, ss = 0.f;
#pragma unroll
            for (int i = 0; i < 32; ++i) {
                float v = sH[rowLN * SH_STRIDE + 4 * i + tLN];
                s += v; ss += v * v;
            }
            s  += __shfl_xor(s, 1);  s  += __shfl_xor(s, 2);
            ss += __shfl_xor(ss, 1); ss += __shfl_xor(ss, 2);
            float mean = s * (1.f / 128.f);
            float var  = ss * (1.f / 128.f) - mean * mean;
            float rs   = rsqrtf(var + 1e-5f);
#pragma unroll
            for (int i = 0; i < 32; ++i) {
                int c = 4 * i + tLN;
                float v  = sH[rowLN * SH_STRIDE + c];
                float xn = (v - mean) * rs * sV[4 * 128 + c] + sV[5 * 128 + c];
                sH[rowLN * SH_STRIDE + c] = 0.5f * xn * (1.f + erff(xn * 0.70710678118f));
            }
        }
        __syncthreads();

        // agg += w_f * (gelu2 + 0.1 * (res + br))   (register accumulation)
        {
            float wf  = __expf(att[f] - amax) / asum;
            float brv = sV[6 * 128 + n];
#pragma unroll
            for (int mt = 0; mt < 4; ++mt)
#pragma unroll
                for (int r = 0; r < 8; ++r) {
                    float hval = sH[(mt * 16 + hf * 8 + r) * SH_STRIDE + n];
                    agg[mt][r] += wf * (hval + 0.1f * (cr[mt][r] + brv));
                }
        }
    }

    __syncthreads();
    // agg + bias -> sH
    {
        float bv = bias[n];
#pragma unroll
        for (int mt = 0; mt < 4; ++mt)
#pragma unroll
            for (int r = 0; r < 8; ++r)
                sH[(mt * 16 + hf * 8 + r) * SH_STRIDE + n] = agg[mt][r] + bv;
    }
    __syncthreads();

    // mixture-beta head: 4 lanes per row
    {
        float dpi[KK] = {0, 0, 0}, da[KK] = {0, 0, 0}, db[KK] = {0, 0, 0};
#pragma unroll
        for (int i = 0; i < 32; ++i) {
            int c = 4 * i + tLN;
            float a = sH[rowLN * SH_STRIDE + c];
#pragma unroll
            for (int k = 0; k < KK; ++k) {
                dpi[k] += a * Wpi[c * KK + k];
                da[k]  += a * Wa [c * KK + k];
                db[k]  += a * Wb [c * KK + k];
            }
        }
#pragma unroll
        for (int k = 0; k < KK; ++k) {
            dpi[k] += __shfl_xor(dpi[k], 1); dpi[k] += __shfl_xor(dpi[k], 2);
            da[k]  += __shfl_xor(da[k], 1);  da[k]  += __shfl_xor(da[k], 2);
            db[k]  += __shfl_xor(db[k], 1);  db[k]  += __shfl_xor(db[k], 2);
        }
        if (tLN == 0) {
            float lg[KK], m = -1e30f;
#pragma unroll
            for (int k = 0; k < KK; ++k) { lg[k] = dpi[k] + bpi[k]; m = fmaxf(m, lg[k]); }
            float se = 0.f;
#pragma unroll
            for (int k = 0; k < KK; ++k) { lg[k] = __expf(lg[k] - m); se += lg[k]; }
            float pred = 0.f;
#pragma unroll
            for (int k = 0; k < KK; ++k) {
                float al = fminf(fmaxf(softplusf(da[k] + ba[k]) + 1.01f, 1.01f), 100.f);
                float bt = fminf(fmaxf(softplusf(db[k] + bb[k]) + 1.01f, 1.01f), 100.f);
                pred += (lg[k] / se) * (al / (al + bt));
            }
            out[b0 + rowLN] = fminf(fmaxf(pred, 0.001f), 0.999f);
        }
    }
}

extern "C" void kernel_launch(void* const* d_in, const int* in_sizes, int n_in,
                              void* d_out, int out_size, void* d_ws, size_t ws_size,
                              hipStream_t stream) {
    const float* x       = (const float*)d_in[0];
    const float* centers = (const float*)d_in[1];
    const float* logw    = (const float*)d_in[2];
    const float* W1      = (const float*)d_in[3];
    const float* b1      = (const float*)d_in[4];
    const float* g1      = (const float*)d_in[5];
    const float* be1     = (const float*)d_in[6];
    const float* W2      = (const float*)d_in[7];
    const float* b2      = (const float*)d_in[8];
    const float* g2      = (const float*)d_in[9];
    const float* be2     = (const float*)d_in[10];
    const float* Wr      = (const float*)d_in[11];
    const float* br      = (const float*)d_in[12];
    const float* att     = (const float*)d_in[13];
    const float* bias    = (const float*)d_in[14];
    const float* Wpi     = (const float*)d_in[15];
    const float* bpi     = (const float*)d_in[16];
    const float* Wa      = (const float*)d_in[17];
    const float* ba      = (const float*)d_in[18];
    const float* Wb      = (const float*)d_in[19];
    const float* bb      = (const float*)d_in[20];

    unsigned short* wsb = (unsigned short*)d_ws;
    const int N1 = FF * NB * H1;           // 524288
    const int N2 = FF * H1 * H2;           // 1048576
    const int NR = FF * NB * H2;           // 524288
    const int NT = N1 + N2 + NR;           // 2097152
    float* rw = (float*)(wsb + NT);        // 16KB reciprocal-width table

    const int cvtN = NT + FF * NB;         // weights + width table
    cvt_weights<<<(cvtN + 255) / 256, 256, 0, stream>>>(W1, W2, Wr, logw, wsb, rw);
    nam_main<<<BB / BM, 256, 0, stream>>>(
        x, centers, rw, b1, g1, be1, b2, g2, be2, br, att, bias,
        Wpi, bpi, Wa, ba, Wb, bb,
        wsb, wsb + N1, wsb + N1 + N2, (float*)d_out);
}
